// HawkesLSTM_21981642621150
// MI455X (gfx1250) — matrix operand order repaired and checked by reading, not compiled, not run on hardware
//
#include <hip/hip_runtime.h>
#include <hip/hip_bf16.h>

#define TT 128          // sequence length
#define BB 128          // batch
#define HH 1024         // hidden
#define PP 32           // process dim
#define KK 1056         // P + H
#define GH (7*HH)       // 7168 fused gate width
#define NWG 64          // one WG per 16 hidden units
#define NTHREADS 256    // 8 waves
#define NKK 33          // K tiles of 32

typedef __attribute__((ext_vector_type(16))) __bf16 v16bf;
typedef __attribute__((ext_vector_type(8)))  __bf16 v8bf;
typedef __attribute__((ext_vector_type(8)))  float  v8f;

static __device__ __forceinline__ float sigmoid_f(float x) {
    return 1.f / (1.f + __expf(-x));
}
static __device__ __forceinline__ float tanh_f(float x) {
    float e = __expf(2.f * x);
    return 1.f - 2.f / (e + 1.f);
}
// branchless, overflow-safe: softplus(x) = max(x,0) + log1p(exp(-|x|))
static __device__ __forceinline__ float softplus_f(float x) {
    return fmaxf(x, 0.f) + __logf(1.f + __expf(-fabsf(x)));
}

static __device__ __forceinline__ v16bf combine16(v8bf lo, v8bf hi) {
    v16bf r;
#pragma unroll
    for (int i = 0; i < 8; ++i) { r[i] = lo[i]; r[i + 8] = hi[i]; }
    return r;
}

// ---------------- prep kernels ----------------

// zero the grid-barrier counter; convert h0 -> bf16 h buffer[0]
__global__ void prep_init(const float* __restrict__ h0,
                          __bf16* __restrict__ hbuf,
                          unsigned* __restrict__ counter) {
    int idx = blockIdx.x * blockDim.x + threadIdx.x;
    if (idx == 0) *counter = 0u;
    if (idx < BB * HH) hbuf[idx] = (__bf16)h0[idx];
}

// W [KK][GH] fp32 -> Wt [GH][KK] bf16 (column-major for B-fragment loads)
__global__ void prep_transpose(const float* __restrict__ W,
                               __bf16* __restrict__ Wt) {
    __shared__ float tile[16][17];
    int kb = blockIdx.x * 16;           // over KK (66)
    int cb = blockIdx.y * 16;           // over GH (448)
    int tx = threadIdx.x & 15, ty = threadIdx.x >> 4;
    tile[ty][tx] = W[(size_t)(kb + ty) * GH + cb + tx];
    __syncthreads();
    Wt[(size_t)(cb + ty) * KK + kb + tx] = (__bf16)tile[tx][ty];
}

// x_all[t][b][p] = bf16(embed[types[t][b]][p])
__global__ void prep_gather_x(const float* __restrict__ embed,
                              const int* __restrict__ types,
                              __bf16* __restrict__ x_all) {
    int idx = blockIdx.x * blockDim.x + threadIdx.x;
    if (idx >= TT * BB * PP) return;
    int p  = idx & (PP - 1);
    int tb = idx >> 5;
    int ty = types[tb];
    x_all[idx] = (__bf16)embed[ty * PP + p];
}

// ---------------- persistent recurrent kernel ----------------
// Block b owns hidden units j0 = b*16 .. +15 (all 7 gates).
// Wave w owns batch rows m0 = w*16 .. +15. c/c_tgt live in registers.

__global__ __launch_bounds__(NTHREADS, 1)
void hawkes_persistent(const float* __restrict__ seq_dt,   // [T][B]
                       const float* __restrict__ c0,       // [B][H]
                       const float* __restrict__ bias,     // [7H]
                       const __bf16* __restrict__ Wt,      // [7H][KK]
                       const __bf16* __restrict__ x_all,   // [T][B][P]
                       __bf16* __restrict__ hbuf,          // [2][B][H]
                       unsigned* __restrict__ counter,
                       float* __restrict__ out) {          // 6 x [T][B][H]
    extern __shared__ __bf16 lds[];     // 7*33 fragments * 512 bf16 = 231 KB

    const int tid   = threadIdx.x;
    const int lane  = tid & 31;
    const int wave  = tid >> 5;
    const int lcol  = lane & 15;
    const int lhalf = (lane < 16) ? 0 : 1;
    const int j0    = blockIdx.x * 16;
    const int m0    = wave * 16;

    // ---- preload all B fragments (W columns, bf16) into LDS once ----
    for (int f = wave; f < 7 * NKK; f += 8) {
        int g  = f / NKK;
        int kk = f % NKK;
        int cg = g * HH + j0 + lcol;                       // gate column
        const __bf16* src = Wt + (size_t)cg * KK + kk * 32 + lhalf * 8;
        __bf16* dst = lds + (size_t)f * 512 + lane * 16;
        *(v8bf*)(dst)     = *(const v8bf*)(src);
        *(v8bf*)(dst + 8) = *(const v8bf*)(src + 16);
    }
    __syncthreads();

    // ---- persistent cell state (tile layout of WMMA C/D) ----
    v8f c, ct;
#pragma unroll
    for (int r = 0; r < 8; ++r) {
        int row = m0 + r + lhalf * 8;
        float v = c0[(size_t)row * HH + j0 + lcol];
        c[r] = v; ct[r] = v;                               // carry = (h0, c0, c0)
    }

    const size_t TBH = (size_t)TT * BB * HH;
    float* out_h  = out;
    float* out_hd = out + TBH;
    float* out_o  = out + 2 * TBH;
    float* out_c  = out + 3 * TBH;
    float* out_ct = out + 4 * TBH;
    float* out_d  = out + 5 * TBH;

    const int arow = m0 + lcol;                            // A-fragment row

#pragma unroll 1
    for (int t = 0; t < TT; ++t) {
        // accumulators init = bias broadcast (C layout: col depends on lane only)
        v8f acc[7];
#pragma unroll
        for (int g = 0; g < 7; ++g) {
            float bg = bias[g * HH + j0 + lcol];
#pragma unroll
            for (int r = 0; r < 8; ++r) acc[g][r] = bg;
        }

        const __bf16* hread = hbuf + (size_t)(t & 1) * BB * HH;
        const __bf16* xrow  = x_all + (size_t)t * BB * PP;

        // ---- peeled K-step 0: embedding part of V ----
        {
            const __bf16* ap = xrow + (size_t)arow * PP + lhalf * 8;
            v16bf a = combine16(*(const v8bf*)ap, *(const v8bf*)(ap + 16));
            const __bf16* lf = lds + lane * 16;
#pragma unroll
            for (int g = 0; g < 7; ++g) {
                const __bf16* bp = lf + (size_t)g * NKK * 512;
                v16bf bm = combine16(*(const v8bf*)bp, *(const v8bf*)(bp + 8));
                acc[g] = __builtin_amdgcn_wmma_f32_16x16x32_bf16(
                    false, a, false, bm, (short)0, acc[g], false, false);
            }
        }

        // ---- K-steps 1..32: recurrent part (unrolled x2 for pipelining) ----
        const __bf16* hrow = hread + (size_t)arow * HH + lhalf * 8;
#pragma unroll 2
        for (int kk = 1; kk < NKK; ++kk) {
            const __bf16* ap = hrow + (kk - 1) * 32;
            v16bf a = combine16(*(const v8bf*)ap, *(const v8bf*)(ap + 16));
            const __bf16* lf = lds + (size_t)kk * 512 + lane * 16;
#pragma unroll
            for (int g = 0; g < 7; ++g) {
                const __bf16* bp = lf + (size_t)g * NKK * 512;
                v16bf bm = combine16(*(const v8bf*)bp, *(const v8bf*)(bp + 8));
                acc[g] = __builtin_amdgcn_wmma_f32_16x16x32_bf16(
                    false, a, false, bm, (short)0, acc[g], false, false);
            }
        }

        // ---- fused elementwise gating in accumulator layout ----
        __bf16* hwrite = hbuf + (size_t)((t + 1) & 1) * BB * HH;
#pragma unroll
        for (int r = 0; r < 8; ++r) {
            int row  = m0 + r + lhalf * 8;
            float dt = seq_dt[t * BB + row];
            float inpt   = sigmoid_f(acc[0][r]);
            float forget = sigmoid_f(acc[1][r]);
            float outp   = sigmoid_f(acc[2][r]);
            float intg   = sigmoid_f(acc[3][r]);
            float fgt    = sigmoid_f(acc[4][r]);
            float z      = tanh_f(acc[5][r]);
            float decay  = softplus_f(8.f * acc[6][r]) * 0.125f;

            float c_i   = forget * c[r] + inpt * z;
            float h_i   = outp * tanh_f(c_i);
            float ctn   = fgt * ct[r] + intg * z;
            float c_new = ctn + (c_i - ctn) * __expf(-decay * dt);
            float h_new = outp * tanh_f(c_new);
            c[r] = c_new; ct[r] = ctn;

            size_t o = (size_t)t * BB * HH + (size_t)row * HH + j0 + lcol;
            out_h[o]  = h_i;
            out_hd[o] = h_new;
            out_o[o]  = outp;
            out_c[o]  = c_i;
            out_ct[o] = ctn;
            out_d[o]  = decay;
            hwrite[(size_t)row * HH + j0 + lcol] = (__bf16)h_new;
        }

        // ---- grid-wide barrier (h_{t+1} complete before next GEMM) ----
        __syncthreads();
        if (tid == 0) {
            __threadfence();
            atomicAdd(counter, 1u);
            unsigned target = (unsigned)(t + 1) * NWG;
            while (*(volatile unsigned*)counter < target) {
                __builtin_amdgcn_s_sleep(1);
            }
        }
        __syncthreads();
    }
}

// ---------------- host launch ----------------

extern "C" void kernel_launch(void* const* d_in, const int* in_sizes, int n_in,
                              void* d_out, int out_size, void* d_ws, size_t ws_size,
                              hipStream_t stream) {
    const float* seq_dt    = (const float*)d_in[0];
    const int*   seq_types = (const int*)d_in[1];
    const float* h0        = (const float*)d_in[2];
    const float* c0        = (const float*)d_in[3];
    const float* embed     = (const float*)d_in[4];
    const float* W         = (const float*)d_in[5];
    const float* b         = (const float*)d_in[6];
    float* out = (float*)d_out;

    // workspace layout
    char* ws = (char*)d_ws;
    unsigned* counter = (unsigned*)ws;                              // 4 B
    __bf16*   hbuf    = (__bf16*)(ws + 256);                        // 2*B*H*2   = 512 KB
    __bf16*   x_all   = (__bf16*)(ws + 256 + 524288);               // T*B*P*2   = 1 MB
    __bf16*   Wt      = (__bf16*)(ws + 256 + 524288 + 1048576);     // GH*KK*2   = ~15 MB
    (void)in_sizes; (void)n_in; (void)out_size; (void)ws_size;

    size_t ldsBytes = (size_t)7 * NKK * 512 * sizeof(__bf16);       // 236544 B
    (void)hipFuncSetAttribute((const void*)hawkes_persistent,
                              hipFuncAttributeMaxDynamicSharedMemorySize,
                              (int)ldsBytes);

    prep_init<<<(BB * HH + 255) / 256, 256, 0, stream>>>(h0, hbuf, counter);
    dim3 tg(KK / 16, GH / 16);                                      // (66, 448)
    prep_transpose<<<tg, 256, 0, stream>>>(W, Wt);
    prep_gather_x<<<(TT * BB * PP + 255) / 256, 256, 0, stream>>>(embed, seq_types, x_all);

    hawkes_persistent<<<NWG, NTHREADS, ldsBytes, stream>>>(
        seq_dt, c0, b, Wt, x_all, hbuf, counter, out);
}
